// SkTWithDecoder_33895881900071
// MI455X (gfx1250) — matrix operand order repaired and checked by reading, not compiled, not run on hardware
//
#include <hip/hip_runtime.h>
#include <hip/hip_bf16.h>

typedef __attribute__((ext_vector_type(16))) _Float16 v16h;
typedef __attribute__((ext_vector_type(8)))  _Float16 v8h;
typedef __attribute__((ext_vector_type(8)))  float    v8f;

#define WMMA_F16(a,b,c) __builtin_amdgcn_wmma_f32_16x16x32_f16(false,(a),false,(b),(short)0,(c),false,false)

__device__ __forceinline__ v16h cat8(v8h lo, v8h hi) {
  return __builtin_shufflevector(lo, hi, 0,1,2,3,4,5,6,7,8,9,10,11,12,13,14,15);
}

// A-matrix 16x32 f16 fragment from row-major source (ld halves, 16B aligned rows).
// lane<16: row=lane, halves 0..15 -> K=0..15
// lane>=16: row=lane-16, halves 0..15 -> K=16..31
__device__ __forceinline__ v16h frag_a_ld(const _Float16* __restrict__ s, int ld) {
  int lane = threadIdx.x & 31;
  const _Float16* p = s + (size_t)(lane & 15) * ld + ((lane < 16) ? 0 : 16);
  v8h lo = *(const v8h*)p;
  v8h hi = *(const v8h*)(p + 8);
  return cat8(lo, hi);
}

// B-matrix 32x16 fragment from column-major source (s[n*32 + k], 16B aligned).
// lane<16: col n=lane, K=0..15 ; lane>=16: col n=lane-16, K=16..31
__device__ __forceinline__ v16h frag_bT32(const _Float16* __restrict__ s) {
  int lane = threadIdx.x & 31;
  const _Float16* p = s + (size_t)(lane & 15) * 32 + ((lane < 16) ? 0 : 16);
  v8h lo = *(const v8h*)p;
  v8h hi = *(const v8h*)(p + 8);
  return cat8(lo, hi);
}

// ---------------- weight convert f32 -> f16, transposed to col-major [n*K + k] ----------------
__global__ __launch_bounds__(256) void k_cvt_t(const float* __restrict__ in,
                                               _Float16* __restrict__ out,
                                               int K, int N, int total) {
  int i = blockIdx.x * 256 + threadIdx.x;
  if (i >= total) return;
  int kn = K * N;
  int mat = i / kn;
  int r = i - mat * kn;
  int k = r / N, n = r - k * N;
  out[(size_t)mat * kn + (size_t)n * K + k] = (_Float16)in[i];
}

// ---------------- patch embed + positional ----------------
__global__ __launch_bounds__(256) void k_patch(const float* __restrict__ x,
                                               const float* __restrict__ cw,
                                               const float* __restrict__ cb,
                                               const float* __restrict__ sp,
                                               const float* __restrict__ tp,
                                               float* __restrict__ tok) {
  int idx = blockIdx.x * 256 + threadIdx.x;
  if (idx >= 16 * 750 * 256) return;
  int e = idx & 255;
  int l = (idx >> 8) % 750;
  int b = idx / (750 * 256);
  int t = l / 25, vv = l % 25;
  float acc = cb[e];
#pragma unroll
  for (int s = 0; s < 4; ++s)
#pragma unroll
    for (int c = 0; c < 3; ++c)
      acc += x[((((size_t)b * 30 + t) * 4 + s) * 25 + vv) * 3 + c] * cw[e * 12 + c * 4 + s];
  acc += tp[e * 30 + t] + sp[e * 25 + vv];
  tok[idx] = acc;
}

// ---------------- stable argsort (rank) + mask, noise row cached in LDS ----------------
__global__ __launch_bounds__(256) void k_argsort(const float* __restrict__ noise,
                                                 int* __restrict__ shuf,
                                                 int* __restrict__ rest,
                                                 float* __restrict__ mask_out) {
  __shared__ float ns[750];
  int b = blockIdx.x;
  const float* nr = noise + (size_t)b * 750;
  for (int i = threadIdx.x; i < 750; i += 256) ns[i] = nr[i];
  __syncthreads();
  for (int i = threadIdx.x; i < 750; i += 256) {
    float vi = ns[i];
    int rank = 0;
    for (int j = 0; j < 750; ++j) {
      float vj = ns[j];
      rank += (vj < vi) || (vj == vi && j < i);
    }
    shuf[b * 750 + rank] = i;
    rest[b * 750 + i] = rank;
    mask_out[b * 750 + i] = (rank < 75) ? 0.0f : 1.0f;
  }
}

// ---------------- gather visible tokens into padded encoder buffer ----------------
__global__ __launch_bounds__(256) void k_gather(const float* __restrict__ tok,
                                                const int* __restrict__ shuf,
                                                float* __restrict__ xe) {
  int idx = blockIdx.x * 256 + threadIdx.x; // 16*80*256
  int e = idx & 255;
  int j = (idx >> 8) % 80;
  int b = idx / (80 * 256);
  float val = 0.0f;
  if (j < 75) {
    int src = shuf[b * 750 + j];
    val = tok[((size_t)b * 750 + src) * 256 + e];
  }
  xe[idx] = val;
}

// ---------------- decoder input build: unshuffle + mask token + dec pos ----------------
__global__ __launch_bounds__(256) void k_decbuild(const float* __restrict__ emb,
                                                  const int* __restrict__ rest,
                                                  const float* __restrict__ mtok,
                                                  const float* __restrict__ dsp,
                                                  const float* __restrict__ dtp,
                                                  float* __restrict__ xd) {
  int idx = blockIdx.x * 256 + threadIdx.x; // 16*752*256
  int e = idx & 255;
  int l = (idx >> 8) % 752;
  int b = idx / (752 * 256);
  float val = 0.0f;
  if (l < 750) {
    int r = rest[b * 750 + l];
    val = (r < 75) ? emb[((size_t)b * 80 + r) * 256 + e] : mtok[e];
    int t = l / 25, vv = l % 25;
    val += dtp[e * 30 + t] + dsp[e * 25 + vv];
  }
  xd[idx] = val;
}

// ---------------- LayerNorm (E=256), one wave per row, vectorized, f16 out ----------------
__global__ __launch_bounds__(256) void k_layernorm(const float* __restrict__ x,
                                                   const float* __restrict__ g,
                                                   const float* __restrict__ bta,
                                                   _Float16* __restrict__ out, int rows) {
  int wid = threadIdx.x >> 5;
  int lane = threadIdx.x & 31;
  int row = blockIdx.x * 8 + wid;
  if (row >= rows) return;
  const float* xr = x + (size_t)row * 256 + lane * 8;
  float4 va = *(const float4*)xr;
  float4 vb = *(const float4*)(xr + 4);
  float s = va.x + va.y + va.z + va.w + vb.x + vb.y + vb.z + vb.w;
#pragma unroll
  for (int off = 16; off > 0; off >>= 1) s += __shfl_xor(s, off, 32);
  float mean = s * (1.0f / 256.0f);
  float vals[8] = {va.x, va.y, va.z, va.w, vb.x, vb.y, vb.z, vb.w};
  float vsum = 0.0f;
#pragma unroll
  for (int j = 0; j < 8; ++j) { float d = vals[j] - mean; vsum += d * d; }
#pragma unroll
  for (int off = 16; off > 0; off >>= 1) vsum += __shfl_xor(vsum, off, 32);
  float inv = rsqrtf(vsum * (1.0f / 256.0f) + 1e-6f);
  const float* gp = g + lane * 8;
  const float* bp = bta + lane * 8;
  float4 g0 = *(const float4*)gp, g1 = *(const float4*)(gp + 4);
  float4 b0 = *(const float4*)bp, b1 = *(const float4*)(bp + 4);
  float gs[8] = {g0.x, g0.y, g0.z, g0.w, g1.x, g1.y, g1.z, g1.w};
  float bs[8] = {b0.x, b0.y, b0.z, b0.w, b1.x, b1.y, b1.z, b1.w};
  v8h o8;
#pragma unroll
  for (int j = 0; j < 8; ++j) o8[j] = (_Float16)((vals[j] - mean) * inv * gs[j] + bs[j]);
  *(v8h*)(out + (size_t)row * 256 + lane * 8) = o8;
}

// ---------------- WMMA GEMM: block tile 128x64, wave tile 32x32 (4 wmma/K-step) ----------------
// A row-major f16 [Mtot x K] (lda), B col-major f16 [n*K + k].
// mode 0: f16 out=(acc+bias)*scale  1: f32 out+=acc+bias  2: f16 out=gelu(acc+bias)
// mode 3: compact f32 pred store     4: f32 out=acc+bias
__global__ __launch_bounds__(256) void k_gemm(const _Float16* __restrict__ A, int lda,
                                              int npad, int lreal,
                                              const _Float16* __restrict__ BT, int K,
                                              const float* __restrict__ bias,
                                              float* __restrict__ outf,
                                              _Float16* __restrict__ outh,
                                              int N, int mode, float scale) {
  __shared__ __align__(16) _Float16 As[128 * 32];
  __shared__ __align__(16) _Float16 Bs[64 * 32];
  const int tid = threadIdx.x;
  const int wid = tid >> 5;
  const int m_off = (wid & 3) * 32;
  const int n_off = (wid >> 2) * 32;
  const int mblk = blockIdx.x * 128;
  const int nblk = blockIdx.y * 64;
  const int ar = tid >> 1, ac = (tid & 1) * 16;   // A: 128 rows x 2 chunks of 16 halves
  const int bcol = tid >> 2, bkc = (tid & 3) * 8; // B: 64 cols x 4 chunks of 8 halves
  const int arow = mblk + ar;
  const bool avalid = (arow % npad) < lreal;
  const _Float16* arp = A + (size_t)arow * lda + ac;
  const _Float16* btp = BT + (size_t)(nblk + bcol) * K + bkc;
  v8f acc[2][2] = {};
  for (int k0 = 0; k0 < K; k0 += 32) {
    uint4 a0 = {0, 0, 0, 0}, a1 = {0, 0, 0, 0};
    if (avalid) {
      a0 = *(const uint4*)(arp + k0);
      a1 = *(const uint4*)(arp + k0 + 8);
    }
    *(uint4*)&As[ar * 32 + ac]     = a0;
    *(uint4*)&As[ar * 32 + ac + 8] = a1;
    *(uint4*)&Bs[bcol * 32 + bkc]  = *(const uint4*)(btp + k0);
    __syncthreads();
    v16h afr[2], bfr[2];
    afr[0] = frag_a_ld(&As[m_off * 32], 32);
    afr[1] = frag_a_ld(&As[(m_off + 16) * 32], 32);
    bfr[0] = frag_bT32(&Bs[n_off * 32]);
    bfr[1] = frag_bT32(&Bs[(n_off + 16) * 32]);
#pragma unroll
    for (int i = 0; i < 2; ++i)
#pragma unroll
      for (int j = 0; j < 2; ++j)
        acc[i][j] = WMMA_F16(afr[i], bfr[j], acc[i][j]);
    __syncthreads();
  }
  const int lane = tid & 31;
  const int cn = lane & 15;
  const int mb = (lane < 16) ? 0 : 8;
#pragma unroll
  for (int i = 0; i < 2; ++i) {
#pragma unroll
    for (int j = 0; j < 2; ++j) {
      const int gn = nblk + n_off + j * 16 + cn;
      const float bv = bias ? bias[gn] : 0.0f;
#pragma unroll
      for (int r = 0; r < 8; ++r) {
        const int gm = mblk + m_off + i * 16 + mb + r;
        float v = acc[i][j][r] + bv;
        if (mode == 0) {
          outh[(size_t)gm * N + gn] = (_Float16)(v * scale);
        } else if (mode == 1) {
          outf[(size_t)gm * N + gn] += v;
        } else if (mode == 2) {
          float gl = 0.5f * v * (1.0f + erff(v * 0.7071067811865476f));
          outh[(size_t)gm * N + gn] = (_Float16)gl;
        } else if (mode == 3) {
          int l = gm % npad;
          if (l < lreal) {
            int b = gm / npad;
            outf[((size_t)b * lreal + l) * N + gn] = v;
          }
        } else {
          outf[(size_t)gm * N + gn] = v;
        }
      }
    }
  }
}

// ---------------- fused flash attention, one wave per (qtile, head, batch) ----------------
__global__ __launch_bounds__(32) void k_attn(const _Float16* __restrict__ q,
                                             const _Float16* __restrict__ k,
                                             const _Float16* __restrict__ v,
                                             _Float16* __restrict__ o,
                                             int npad, int lreal) {
  __shared__ __align__(16) _Float16 P[16 * 32];
  __shared__ __align__(16) _Float16 Vt[32 * 32]; // [dim 0..31][token 0..31]
  const int qt = blockIdx.x, h = blockIdx.y, b = blockIdx.z;
  const int lane = threadIdx.x;
  const int n = lane & 15;
  const int mb = (lane < 16) ? 0 : 8;
  const int kb = (lane < 16) ? 0 : 16;
  const size_t base = (size_t)b * npad;
  const int hoff = h * 32;

  v16h qa = frag_a_ld(q + (base + qt * 16) * 256 + hoff, 256);

  v8f acc0 = {}, acc1 = {};
  float mrow[8], lrow[8];
#pragma unroll
  for (int r = 0; r < 8; ++r) { mrow[r] = -1e30f; lrow[r] = 0.0f; }

  for (int kt = 0; kt < npad; kt += 32) {
    // ---- stage V tile (32 tokens x 32 dims) transposed into LDS ----
    int tv = kt + lane; if (tv > npad - 1) tv = npad - 1;
    {
      const _Float16* vp = v + (base + tv) * 256 + hoff;
      v8h c0 = *(const v8h*)vp;
      v8h c1 = *(const v8h*)(vp + 8);
      v8h c2 = *(const v8h*)(vp + 16);
      v8h c3 = *(const v8h*)(vp + 24);
#pragma unroll
      for (int d = 0; d < 8; ++d) {
        Vt[d * 32 + lane]        = c0[d];
        Vt[(8 + d) * 32 + lane]  = c1[d];
        Vt[(16 + d) * 32 + lane] = c2[d];
        Vt[(24 + d) * 32 + lane] = c3[d];
      }
    }
    // ---- S = Q*K^T for 32 keys (two 16-key tiles), contiguous 32B per lane ----
    int t0 = kt + n;      if (t0 > npad - 1) t0 = npad - 1;
    int t1 = kt + 16 + n; if (t1 > npad - 1) t1 = npad - 1;
    const _Float16* kp0 = k + (base + t0) * 256 + hoff + kb;
    const _Float16* kp1 = k + (base + t1) * 256 + hoff + kb;
    v16h kf0 = cat8(*(const v8h*)kp0, *(const v8h*)(kp0 + 8));
    v16h kf1 = cat8(*(const v8h*)kp1, *(const v8h*)(kp1 + 8));
    v8f zero = {};
    v8f s0 = WMMA_F16(qa, kf0, zero);
    v8f s1 = WMMA_F16(qa, kf1, zero);
    const bool msk0 = (kt + n) >= lreal;
    const bool msk1 = (kt + 16 + n) >= lreal;
#pragma unroll
    for (int r = 0; r < 8; ++r) {
      if (msk0) s0[r] = -1e30f;
      if (msk1) s1[r] = -1e30f;
    }
    // ---- online softmax (row reductions across 16-lane half) ----
    float p0[8], p1[8];
#pragma unroll
    for (int r = 0; r < 8; ++r) {
      float tmax = fmaxf(s0[r], s1[r]);
#pragma unroll
      for (int off = 8; off > 0; off >>= 1) tmax = fmaxf(tmax, __shfl_xor(tmax, off, 32));
      float mnew = fmaxf(mrow[r], tmax);
      float corr = __expf(mrow[r] - mnew);
      float e0 = __expf(s0[r] - mnew);
      float e1 = __expf(s1[r] - mnew);
      float rs = e0 + e1;
#pragma unroll
      for (int off = 8; off > 0; off >>= 1) rs += __shfl_xor(rs, off, 32);
      lrow[r] = lrow[r] * corr + rs;
      mrow[r] = mnew;
      acc0[r] *= corr; acc1[r] *= corr;
      p0[r] = e0; p1[r] = e1;
    }
    // ---- stage P, reload as A-frag, O += P*V via LDS-transposed V ----
#pragma unroll
    for (int r = 0; r < 8; ++r) {
      P[(mb + r) * 32 + n]      = (_Float16)p0[r];
      P[(mb + r) * 32 + 16 + n] = (_Float16)p1[r];
    }
    __syncthreads();
    v16h pa = frag_a_ld(P, 32);
    const _Float16* v0p = &Vt[n * 32 + kb];
    const _Float16* v1p = &Vt[(16 + n) * 32 + kb];
    v16h vf0 = cat8(*(const v8h*)v0p, *(const v8h*)(v0p + 8));
    v16h vf1 = cat8(*(const v8h*)v1p, *(const v8h*)(v1p + 8));
    acc0 = WMMA_F16(pa, vf0, acc0);
    acc1 = WMMA_F16(pa, vf1, acc1);
    __syncthreads();
  }
  _Float16* op = o + (base + qt * 16) * 256 + hoff;
#pragma unroll
  for (int r = 0; r < 8; ++r) {
    float inv = 1.0f / lrow[r];
    op[(size_t)(mb + r) * 256 + n]      = (_Float16)(acc0[r] * inv);
    op[(size_t)(mb + r) * 256 + 16 + n] = (_Float16)(acc1[r] * inv);
  }
}

// ==================================================================================
extern "C" void kernel_launch(void* const* d_in, const int* in_sizes, int n_in,
                              void* d_out, int out_size, void* d_ws, size_t ws_size,
                              hipStream_t stream) {
  (void)in_sizes; (void)n_in; (void)out_size; (void)ws_size;
  const float* x           = (const float*)d_in[0];
  const float* noise       = (const float*)d_in[1];
  const float* conv_w      = (const float*)d_in[2];
  const float* conv_b      = (const float*)d_in[3];
  const float* spatio_pos  = (const float*)d_in[4];
  const float* temporal_pos= (const float*)d_in[5];
  const float* enc_ln_s    = (const float*)d_in[6];
  const float* enc_ln_b    = (const float*)d_in[7];
  const float* enc_attn_w  = (const float*)d_in[8];
  const float* enc_attn_b  = (const float*)d_in[9];
  const float* enc_fc1_w   = (const float*)d_in[10];
  const float* enc_fc1_b   = (const float*)d_in[11];
  const float* enc_fc2_w   = (const float*)d_in[12];
  const float* enc_fc2_b   = (const float*)d_in[13];
  const float* dec_ln_s    = (const float*)d_in[14];
  const float* dec_ln_b    = (const float*)d_in[15];
  const float* dec_attn_w  = (const float*)d_in[16];
  const float* dec_attn_b  = (const float*)d_in[17];
  const float* dec_fc1_w   = (const float*)d_in[18];
  const float* dec_fc1_b   = (const float*)d_in[19];
  const float* dec_fc2_w   = (const float*)d_in[20];
  const float* dec_fc2_b   = (const float*)d_in[21];
  const float* enc_norm_s  = (const float*)d_in[22];
  const float* enc_norm_b  = (const float*)d_in[23];
  const float* dec_embed_w = (const float*)d_in[24];
  const float* dec_embed_b = (const float*)d_in[25];
  const float* mask_token  = (const float*)d_in[26];
  const float* dec_sp      = (const float*)d_in[27];
  const float* dec_tp      = (const float*)d_in[28];
  const float* dec_norm_s  = (const float*)d_in[29];
  const float* dec_norm_b  = (const float*)d_in[30];
  const float* head_w      = (const float*)d_in[31];
  const float* head_b      = (const float*)d_in[32];

  float* pred = (float*)d_out;
  float* mask = pred + (size_t)16 * 750 * 256;

  char* wp = (char*)d_ws;
  auto take = [&](size_t bytes) -> void* {
    void* r = wp; wp += (bytes + 255) & ~(size_t)255; return r;
  };
  float* tok  = (float*)take((size_t)16 * 750 * 256 * 4);
  int*   shuf = (int*)  take((size_t)16 * 750 * 4);
  int*   rest = (int*)  take((size_t)16 * 750 * 4);
  float* xe   = (float*)take((size_t)16 * 80 * 256 * 4);
  float* xd   = (float*)take((size_t)16 * 752 * 256 * 4);
  float* emb  = (float*)take((size_t)16 * 80 * 256 * 4);
  _Float16* hln = (_Float16*)take((size_t)16 * 752 * 256 * 2);
  _Float16* qf  = (_Float16*)take((size_t)16 * 752 * 256 * 2);
  _Float16* kf  = (_Float16*)take((size_t)16 * 752 * 256 * 2);
  _Float16* vf  = (_Float16*)take((size_t)16 * 752 * 256 * 2);
  _Float16* of  = (_Float16*)take((size_t)16 * 752 * 256 * 2);
  _Float16* hid = (_Float16*)take((size_t)16 * 752 * 1024 * 2);
  _Float16* wea = (_Float16*)take((size_t)8 * 4 * 65536 * 2);
  _Float16* we1 = (_Float16*)take((size_t)8 * 262144 * 2);
  _Float16* we2 = (_Float16*)take((size_t)8 * 262144 * 2);
  _Float16* wda = (_Float16*)take((size_t)4 * 4 * 65536 * 2);
  _Float16* wd1 = (_Float16*)take((size_t)4 * 262144 * 2);
  _Float16* wd2 = (_Float16*)take((size_t)4 * 262144 * 2);
  _Float16* wde = (_Float16*)take((size_t)65536 * 2);
  _Float16* whd = (_Float16*)take((size_t)65536 * 2);

  auto cvtT = [&](const float* src, _Float16* dst, int K, int N, size_t total) {
    k_cvt_t<<<dim3((unsigned)((total + 255) / 256)), 256, 0, stream>>>(src, dst, K, N, (int)total);
  };
  auto gemm = [&](const _Float16* A, int lda, int npad, int lreal, int Mtot,
                  const _Float16* BT, int K, const float* bias,
                  float* outf, _Float16* outh, int N, int mode, float scale) {
    dim3 g(Mtot / 128, N / 64);
    k_gemm<<<g, 256, 0, stream>>>(A, lda, npad, lreal, BT, K, bias, outf, outh, N, mode, scale);
  };
  auto ln = [&](const float* xin, const float* g_, const float* b_, _Float16* o_, int rows) {
    k_layernorm<<<dim3(rows / 8), 256, 0, stream>>>(xin, g_, b_, o_, rows);
  };

  // ---- weight conversion to f16, transposed to col-major ----
  cvtT(enc_attn_w, wea, 256, 256,  (size_t)32 * 65536);
  cvtT(enc_fc1_w,  we1, 256, 1024, (size_t)8 * 262144);
  cvtT(enc_fc2_w,  we2, 1024, 256, (size_t)8 * 262144);
  cvtT(dec_attn_w, wda, 256, 256,  (size_t)16 * 65536);
  cvtT(dec_fc1_w,  wd1, 256, 1024, (size_t)4 * 262144);
  cvtT(dec_fc2_w,  wd2, 1024, 256, (size_t)4 * 262144);
  cvtT(dec_embed_w, wde, 256, 256, 65536);
  cvtT(head_w,      whd, 256, 256, 65536);

  // ---- front end ----
  k_patch<<<dim3(12000), 256, 0, stream>>>(x, conv_w, conv_b, spatio_pos, temporal_pos, tok);
  k_argsort<<<dim3(16), 256, 0, stream>>>(noise, shuf, rest, mask);
  k_gather<<<dim3(1280), 256, 0, stream>>>(tok, shuf, xe);

  const float qscale = 0.17677669529663687f; // 1/sqrt(32)

  // ---- encoder: 8 blocks, 16 batches x 80(pad of 75) tokens ----
  for (int i = 0; i < 8; ++i) {
    ln(xe, enc_ln_s + (i * 2 + 0) * 256, enc_ln_b + (i * 2 + 0) * 256, hln, 1280);
    gemm(hln, 256, 80, 75, 1280, wea + (size_t)(i * 4 + 0) * 65536, 256,
         enc_attn_b + (i * 4 + 0) * 256, nullptr, qf, 256, 0, qscale);
    gemm(hln, 256, 80, 75, 1280, wea + (size_t)(i * 4 + 1) * 65536, 256,
         enc_attn_b + (i * 4 + 1) * 256, nullptr, kf, 256, 0, 1.0f);
    gemm(hln, 256, 80, 75, 1280, wea + (size_t)(i * 4 + 2) * 65536, 256,
         enc_attn_b + (i * 4 + 2) * 256, nullptr, vf, 256, 0, 1.0f);
    k_attn<<<dim3(5, 8, 16), 32, 0, stream>>>(qf, kf, vf, of, 80, 75);
    gemm(of, 256, 80, 75, 1280, wea + (size_t)(i * 4 + 3) * 65536, 256,
         enc_attn_b + (i * 4 + 3) * 256, xe, nullptr, 256, 1, 1.0f);
    ln(xe, enc_ln_s + (i * 2 + 1) * 256, enc_ln_b + (i * 2 + 1) * 256, hln, 1280);
    gemm(hln, 256, 80, 75, 1280, we1 + (size_t)i * 262144, 256,
         enc_fc1_b + (size_t)i * 1024, nullptr, hid, 1024, 2, 1.0f);
    gemm(hid, 1024, 80, 75, 1280, we2 + (size_t)i * 262144, 1024,
         enc_fc2_b + (size_t)i * 256, xe, nullptr, 256, 1, 1.0f);
  }

  // ---- encoder norm + decoder embed + unshuffle ----
  ln(xe, enc_norm_s, enc_norm_b, hln, 1280);
  gemm(hln, 256, 80, 75, 1280, wde, 256, dec_embed_b, emb, nullptr, 256, 4, 1.0f);
  k_decbuild<<<dim3(12032), 256, 0, stream>>>(emb, rest, mask_token, dec_sp, dec_tp, xd);

  // ---- decoder: 4 blocks, 16 batches x 752(pad of 750) tokens ----
  for (int i = 0; i < 4; ++i) {
    ln(xd, dec_ln_s + (i * 2 + 0) * 256, dec_ln_b + (i * 2 + 0) * 256, hln, 12032);
    gemm(hln, 256, 752, 750, 12032, wda + (size_t)(i * 4 + 0) * 65536, 256,
         dec_attn_b + (i * 4 + 0) * 256, nullptr, qf, 256, 0, qscale);
    gemm(hln, 256, 752, 750, 12032, wda + (size_t)(i * 4 + 1) * 65536, 256,
         dec_attn_b + (i * 4 + 1) * 256, nullptr, kf, 256, 0, 1.0f);
    gemm(hln, 256, 752, 750, 12032, wda + (size_t)(i * 4 + 2) * 65536, 256,
         dec_attn_b + (i * 4 + 2) * 256, nullptr, vf, 256, 0, 1.0f);
    k_attn<<<dim3(47, 8, 16), 32, 0, stream>>>(qf, kf, vf, of, 752, 750);
    gemm(of, 256, 752, 750, 12032, wda + (size_t)(i * 4 + 3) * 65536, 256,
         dec_attn_b + (i * 4 + 3) * 256, xd, nullptr, 256, 1, 1.0f);
    ln(xd, dec_ln_s + (i * 2 + 1) * 256, dec_ln_b + (i * 2 + 1) * 256, hln, 12032);
    gemm(hln, 256, 752, 750, 12032, wd1 + (size_t)i * 262144, 256,
         dec_fc1_b + (size_t)i * 1024, nullptr, hid, 1024, 2, 1.0f);
    gemm(hid, 1024, 752, 750, 12032, wd2 + (size_t)i * 262144, 1024,
         dec_fc2_b + (size_t)i * 256, xd, nullptr, 256, 1, 1.0f);
  }

  // ---- final norm + head, compact store into d_out ----
  ln(xd, dec_norm_s, dec_norm_b, hln, 12032);
  gemm(hln, 256, 752, 750, 12032, whd, 256, head_b, pred, nullptr, 256, 3, 1.0f);
}
